// CopyGenerator_83133386981903
// MI455X (gfx1250) — matrix operand order repaired and checked by reading, not compiled, not run on hardware
//
#include <hip/hip_runtime.h>
#include <hip/hip_bf16.h>
#include <math.h>

// ---------------------------------------------------------------------------
// CDNA5 / gfx1250 pointer-generator decode step.
// Heavy lifting done with V_WMMA_F32_16X16X32_BF16 (wave32 WMMA).
// ---------------------------------------------------------------------------

typedef __attribute__((ext_vector_type(16))) __bf16 v16bf;
typedef __attribute__((ext_vector_type(8)))  float  v8f;

union WFrag {
    v16bf v;
    uint4 q[2];
};

// Round-to-nearest (ties away) f32->bf16 for both values, then merge the two
// high halves with a single v_perm_b32: result = {bf16(y), bf16(x)}.
__device__ __forceinline__ unsigned int pack_bf16(float x, float y) {
    unsigned int ux = __float_as_uint(x) + 0x8000u;
    unsigned int uy = __float_as_uint(y) + 0x8000u;
    // src0 (uy) supplies bytes 4-7, src1 (ux) supplies bytes 0-3.
    // selector 0x07060302: out = { uy[3], uy[2], ux[3], ux[2] }
    return __builtin_amdgcn_perm(uy, ux, 0x07060302u);
}

// Generic tiled GEMM:  C = A(MxK) * B(KxN) [+bias]  with optional per-row blend.
//   MODE 0: C = acc + bias
//   MODE 1: C = (1 - aCopy[row & 7]) * (acc + bias)     (final vocab projection)
// TRANSB: B element (k,n) read from Bm[n*ldb + k] instead of Bm[k*ldb + n].
// Batched via blockIdx.z with element offsets aBatch/bBatch/cBatch.
// Requires: M%128==0 per grid.x, N%128==0 per grid.y, K%32==0.
template <int TRANSB, int MODE>
__global__ __launch_bounds__(256)
void gemm_wmma_bf16(const float* __restrict__ A, const float* __restrict__ Bm,
                    const float* __restrict__ bias, float* __restrict__ C,
                    const float* __restrict__ aCopy,
                    int K, int lda, int ldb, int ldc,
                    long long aBatch, long long bBatch, long long cBatch) {
    __shared__ __align__(16) unsigned short As[128 * 40]; // bf16, row stride 40
    __shared__ __align__(16) unsigned short Bs[128 * 40]; // bf16, n-major, stride 40

    const int tid  = threadIdx.x;
    const int lane = tid & 31;
    const int w    = tid >> 5;       // 8 waves
    const int wm   = w & 3;          // wave row (2 m-tiles each)
    const int wn   = w >> 2;         // wave col (4 n-tiles each)
    const int m0   = blockIdx.x * 128;
    const int n0   = blockIdx.y * 128;
    const int z    = blockIdx.z;

    const float* Ab = A  + (size_t)z * (size_t)aBatch;
    const float* Bb = Bm + (size_t)z * (size_t)bBatch;
    float*       Cb = C  + (size_t)z * (size_t)cBatch;

    v8f acc[2][4];
#pragma unroll
    for (int mt = 0; mt < 2; ++mt)
#pragma unroll
        for (int nt = 0; nt < 4; ++nt)
            acc[mt][nt] = (v8f){0.f, 0.f, 0.f, 0.f, 0.f, 0.f, 0.f, 0.f};

    const int l15 = lane & 15;
    const int hi  = (lane & 16) ? 1 : 0;

    for (int k0 = 0; k0 < K; k0 += 32) {
        // ---- prefetch next K-chunk (branchless clamp; K % 32 == 0) ----
        {
            const int kn = min(k0 + 32, K - 32);
            const int pr   = tid & 127;
            const int half = (tid >> 7) * 16;      // 0 or 16 floats (64B line)
            __builtin_prefetch(Ab + (size_t)(m0 + pr) * lda + kn + half, 0, 3);
            if (TRANSB) {
                __builtin_prefetch(Bb + (size_t)(n0 + pr) * ldb + kn + half, 0, 3);
            } else {
                const int krp = tid >> 3;          // 0..31 k-rows
                const int seg = (tid & 7) * 16;    // 8 x 64B segments
                __builtin_prefetch(Bb + (size_t)(kn + krp) * ldb + n0 + seg, 0, 3);
            }
        }

        // ---- stage A tile (128 rows x 32 k) as bf16, row stride 40 ----
        {
            const int i0 = tid >> 4;          // 0..15
            const int kp = tid & 15;          // k-pair index
#pragma unroll
            for (int ii = 0; ii < 8; ++ii) {
                const int i = i0 + (ii << 4);
                const float2 f2 = *(const float2*)(Ab + (size_t)(m0 + i) * lda + k0 + 2 * kp);
                ((unsigned int*)As)[i * 20 + kp] = pack_bf16(f2.x, f2.y);
            }
        }
        // ---- stage B tile transposed into Bs[n][k] ----
        if (TRANSB) {
            const int nr = tid >> 4;
            const int kp = tid & 15;
#pragma unroll
            for (int ii = 0; ii < 8; ++ii) {
                const int n = nr + (ii << 4);
                const float2 f2 = *(const float2*)(Bb + (size_t)(n0 + n) * ldb + k0 + 2 * kp);
                ((unsigned int*)Bs)[n * 20 + kp] = pack_bf16(f2.x, f2.y);
            }
        } else {
            const int c  = tid & 127;         // column n
            const int pr = tid >> 7;          // 0/1
#pragma unroll
            for (int kk = 0; kk < 8; ++kk) {
                const int kpair = pr + (kk << 1);      // 0..15
                const int krow  = kpair * 2;
                const float x0 = Bb[(size_t)(k0 + krow)     * ldb + n0 + c];
                const float x1 = Bb[(size_t)(k0 + krow + 1) * ldb + n0 + c];
                ((unsigned int*)Bs)[c * 20 + kpair] = pack_bf16(x0, x1);
            }
        }
        __syncthreads();

        // ---- load fragments and issue WMMAs ----
        WFrag af[2], bfr[4];
        const int kofsA = hi ? 8 : 0;          // A lanes>=16 hold K {8..15,24..31}
#pragma unroll
        for (int mt = 0; mt < 2; ++mt) {
            const int arow = (wm << 5) + (mt << 4) + l15;
            af[mt].q[0] = *(const uint4*)(As + arow * 40 + kofsA);
            af[mt].q[1] = *(const uint4*)(As + arow * 40 + kofsA + 16);
        }
        const int kofsB = hi ? 8 : 0;          // B lanes>=16 hold K {8..15,24..31}
#pragma unroll
        for (int nt = 0; nt < 4; ++nt) {
            const int bn = (wn << 6) + (nt << 4) + l15;
            bfr[nt].q[0] = *(const uint4*)(Bs + bn * 40 + kofsB);
            bfr[nt].q[1] = *(const uint4*)(Bs + bn * 40 + kofsB + 16);
        }
#pragma unroll
        for (int mt = 0; mt < 2; ++mt)
#pragma unroll
            for (int nt = 0; nt < 4; ++nt)
                acc[mt][nt] = __builtin_amdgcn_wmma_f32_16x16x32_bf16(
                    false, af[mt].v, false, bfr[nt].v,
                    (short)0, acc[mt][nt], false, false);
        __syncthreads();
    }

    // ---- write back (C/D layout: VGPR r -> M=r (lanes 0-15) / r+8 (16-31)) ----
#pragma unroll
    for (int mt = 0; mt < 2; ++mt) {
#pragma unroll
        for (int nt = 0; nt < 4; ++nt) {
            const int col = n0 + (wn << 6) + (nt << 4) + l15;
            const float bv = bias ? bias[col] : 0.f;
#pragma unroll
            for (int r = 0; r < 8; ++r) {
                const int row = m0 + (wm << 5) + (mt << 4) + r + (hi ? 8 : 0);
                float v = acc[mt][nt][r] + bv;
                if (MODE == 1) v *= (1.0f - aCopy[row & 7]);
                Cb[(size_t)row * ldc + col] = v;
            }
        }
    }
}

// Softmax over s (128) for each (b,t) row, with 1/sqrt(D) logit scale.
__global__ void softmax_rows(float* __restrict__ attn) {
    __shared__ float red[128];
    const int r = blockIdx.x;                  // b*128 + t
    float* p = attn + (size_t)r * 128;
    const int s = threadIdx.x;
    const float scale = 0.04419417382415922f;  // 1/sqrt(512)
    float v = p[s] * scale;
    red[s] = v; __syncthreads();
    for (int off = 64; off > 0; off >>= 1) {
        if (s < off) red[s] = fmaxf(red[s], red[s + off]);
        __syncthreads();
    }
    const float m = red[0]; __syncthreads();
    const float e = __expf(v - m);
    red[s] = e; __syncthreads();
    for (int off = 64; off > 0; off >>= 1) {
        if (s < off) red[s] += red[s + off];
        __syncthreads();
    }
    p[s] = e / red[0];
}

// a[b] = sigmoid( ((sum_t x[b,t,:]) @ Wf + 128*bf) @ Wc + bc )
__global__ void acopy_kernel(const float* __restrict__ x, const float* __restrict__ Wf,
                             const float* __restrict__ bfv, const float* __restrict__ Wc,
                             const float* __restrict__ bc, float* __restrict__ a) {
    __shared__ float sv[512];
    __shared__ float red[256];
    const int b = blockIdx.x;
    const int t = threadIdx.x;                 // 256 threads
    for (int d = t; d < 512; d += 256) {
        float acc = 0.f;
        const float* xp = x + (size_t)b * 128 * 512 + d;
        for (int tt = 0; tt < 128; ++tt) acc += xp[(size_t)tt * 512];
        sv[d] = acc;
    }
    __syncthreads();
    float part = 0.f;
    for (int d2 = t; d2 < 512; d2 += 256) {
        float zv = 0.f;
        for (int d = 0; d < 512; ++d) zv += sv[d] * Wf[(size_t)d * 512 + d2];
        part += (zv + 128.0f * bfv[d2]) * Wc[d2];
    }
    red[t] = part; __syncthreads();
    for (int off = 128; off > 0; off >>= 1) {
        if (t < off) red[t] += red[t + off];
        __syncthreads();
    }
    if (t == 0) {
        const float y = red[0] + bc[0];
        a[b] = 1.0f / (1.0f + __expf(-y));
    }
}

// out[t,b,src[s,b]] += a[b] * attn[b,t,s]
__global__ void scatter_kernel(const float* __restrict__ attn, const int* __restrict__ src,
                               const float* __restrict__ a, float* __restrict__ out) {
    const int r = blockIdx.x;                  // b*128 + t
    const int b = r >> 7, t = r & 127;
    const int s = threadIdx.x;                 // 128
    const float val = a[b] * attn[(size_t)r * 128 + s];
    const int vi = src[s * 8 + b];
    atomicAdd(out + ((size_t)(t * 8 + b)) * 32000 + vi, val);
}

extern "C" void kernel_launch(void* const* d_in, const int* in_sizes, int n_in,
                              void* d_out, int out_size, void* d_ws, size_t ws_size,
                              hipStream_t stream) {
    (void)in_sizes; (void)n_in; (void)out_size; (void)ws_size;
    const float* htgt = (const float*)d_in[0];   // (128,8,512)
    const float* hsrc = (const float*)d_in[1];   // (128,8,512)
    const int*   src  = (const int*)d_in[2];     // (128,8)
    const float* Wq   = (const float*)d_in[3];   // (512,512)
    const float* bq   = (const float*)d_in[4];   // (512,)
    const float* Wf   = (const float*)d_in[5];   // (512,512)
    const float* bfv  = (const float*)d_in[6];   // (512,)
    const float* Wg   = (const float*)d_in[7];   // (512,32000)
    const float* bg   = (const float*)d_in[8];   // (32000,)
    const float* Wc   = (const float*)d_in[9];   // (512,1)
    const float* bc   = (const float*)d_in[10];  // (1,)
    float* out = (float*)d_out;                  // (128,8,32000)
    float* ws  = (float*)d_ws;

    float* Qp   = ws;                 // 1024x512
    float* Kp   = ws + 524288;        // 1024x512  (k == v, both use Wq)
    float* attn = ws + 1048576;       // 8x128x128
    float* x    = ws + 1179648;       // 8x128x512
    float* a    = ws + 1703936;       // 8

    const dim3 blk(256);

    // Q = htgt@Wq + bq ; K = V = hsrc@Wq + bq   (rows indexed (t*8+b))
    gemm_wmma_bf16<0, 0><<<dim3(8, 4, 1), blk, 0, stream>>>(
        htgt, Wq, bq, Qp, nullptr, 512, 512, 512, 512, 0, 0, 0);
    gemm_wmma_bf16<0, 0><<<dim3(8, 4, 1), blk, 0, stream>>>(
        hsrc, Wq, bq, Kp, nullptr, 512, 512, 512, 512, 0, 0, 0);

    // logits[b,t,s] = q[b,t,:] . k[b,s,:]   (B-transposed, batched over b)
    gemm_wmma_bf16<1, 0><<<dim3(1, 1, 8), blk, 0, stream>>>(
        Qp, Kp, nullptr, attn, nullptr, 512, 4096, 4096, 128,
        512LL, 512LL, 16384LL);

    softmax_rows<<<1024, 128, 0, stream>>>(attn);

    // x[b,t,:] = attn[b,t,:] @ V_b
    gemm_wmma_bf16<0, 0><<<dim3(1, 4, 8), blk, 0, stream>>>(
        attn, Kp, nullptr, x, nullptr, 128, 128, 4096, 512,
        16384LL, 512LL, 65536LL);

    acopy_kernel<<<8, 256, 0, stream>>>(x, Wf, bfv, Wc, bc, a);

    // out = (1-a[b]) * (htgt@Wg + bg)   — dominant GEMM, blended write
    gemm_wmma_bf16<0, 1><<<dim3(8, 250, 1), blk, 0, stream>>>(
        htgt, Wg, bg, out, a, 512, 512, 32000, 32000, 0, 0, 0);

    // out += a[b] * scatter(attn)
    scatter_kernel<<<1024, 128, 0, stream>>>(attn, src, a, out);
}
